// ViTSelfOutput_8117488190190
// MI455X (gfx1250) — matrix operand order repaired and checked by reading, not compiled, not run on hardware
//
#include <hip/hip_runtime.h>

typedef __attribute__((ext_vector_type(16))) _Float16 v16h;
typedef __attribute__((ext_vector_type(8)))  _Float16 v8h;
typedef __attribute__((ext_vector_type(4)))  _Float16 v4h;
typedef __attribute__((ext_vector_type(8)))  float    v8f;
typedef __attribute__((ext_vector_type(4)))  float    v4f;

static constexpr int kH   = 768;   // hidden dim (K and N of the GEMM)
static constexpr int kBM  = 128;   // block tile M
static constexpr int kBN  = 128;   // block tile N
static constexpr int kBK  = 32;    // K step (matches WMMA 16x16x32)
static constexpr int kBKP = 40;    // padded LDS row stride in halves (80B)
static constexpr int kTile = kBM * kBKP;   // halves per LDS tile buffer

static __device__ inline v16h combine16(v8h lo, v8h hi) {
  v16h r;
#pragma unroll
  for (int i = 0; i < 8; ++i) { r[i] = lo[i]; r[i + 8] = hi[i]; }
  return r;
}

// Issue async global->LDS copies for one 128x32 f16 tile (no wait here).
// Each thread moves 2 x 16B chunks, unconditionally. Out-of-range A rows are
// clamped to row_max: their LDS contents are don't-care because the epilogue
// never stores C rows >= M.
static __device__ inline void stage_tile_async(const _Float16* __restrict__ gbase,
                                               int grow0, int row_max, int k0,
                                               _Float16* lds_tile, int tid)
{
#pragma unroll
  for (int i = 0; i < 2; ++i) {
    const int c   = tid + i * 256;   // 16B-chunk id, 0..511
    const int row = c >> 2;          // 0..127
    const int q   = c & 3;           // which 16B chunk of the 64B row
    int gr = grow0 + row;
    gr = (gr > row_max) ? row_max : gr;     // clamp, no EXEC masking
    const _Float16* g = gbase + (size_t)gr * kH + k0 + q * 8;
    const unsigned   l = (unsigned)(uintptr_t)(lds_tile + row * kBKP + q * 8);
    asm volatile("global_load_async_to_lds_b128 %0, %1, off"
                 :: "v"(l), "v"(g) : "memory");
  }
}

static __device__ inline void wait_async_and_barrier() {
  asm volatile("s_wait_asynccnt 0" ::: "memory");
  __syncthreads();
}

// x + move -> f16  (8 elements / thread)
__global__ __launch_bounds__(256) void cvt_x_f16(
    const float* __restrict__ x, const float* __restrict__ move,
    _Float16* __restrict__ xh, int total)
{
  const int i = (blockIdx.x * 256 + threadIdx.x) * 8;
  if (i >= total) return;
  v4f a = *(const v4f*)(x + i);
  v4f b = *(const v4f*)(x + i + 4);
  const int c = i % kH;              // 8 | 768, so c..c+7 stay within one row
  v8h o;
#pragma unroll
  for (int j = 0; j < 4; ++j) {
    o[j]     = (_Float16)(a[j] + move[c + j]);
    o[j + 4] = (_Float16)(b[j] + move[c + 4 + j]);
  }
  *(v8h*)(xh + i) = o;
}

// W -> f16
__global__ __launch_bounds__(256) void cvt_w_f16(
    const float* __restrict__ W, _Float16* __restrict__ Wh, int total)
{
  const int i = (blockIdx.x * 256 + threadIdx.x) * 8;
  if (i >= total) return;
  v4f a = *(const v4f*)(W + i);
  v4f b = *(const v4f*)(W + i + 4);
  v8h o;
#pragma unroll
  for (int j = 0; j < 4; ++j) { o[j] = (_Float16)a[j]; o[j + 4] = (_Float16)b[j]; }
  *(v8h*)(Wh + i) = o;
}

// y[m, o] = sum_h xh[m,h] * Wh[o,h] + bias[o]   (f16 in, f32 out)
__global__ __launch_bounds__(256) void gemm_wmma_f16(
    const _Float16* __restrict__ xh, const _Float16* __restrict__ Wh,
    const float* __restrict__ bias, float* __restrict__ y, int M)
{
  __shared__ __align__(16) _Float16 As[2][kTile];
  __shared__ __align__(16) _Float16 Bs[2][kTile];

  const int tid  = threadIdx.x;
  const int lane = tid & 31;
  const int wave = tid >> 5;
  const int wm   = wave >> 1;        // 0..3 -> 32 rows each
  const int wn   = wave & 1;         // 0..1 -> 64 cols each
  const int bm   = blockIdx.y * kBM;
  const int bn   = blockIdx.x * kBN;
  const int hi   = lane >> 4;        // 0/1
  const int lo   = lane & 15;
  const int rmax = M - 1;

  v8f acc[2][4];
#pragma unroll
  for (int mt = 0; mt < 2; ++mt)
#pragma unroll
    for (int nt = 0; nt < 4; ++nt)
#pragma unroll
      for (int r = 0; r < 8; ++r) acc[mt][nt][r] = 0.0f;

  // Prologue: fill buffer 0
  stage_tile_async(xh, bm, rmax,    0, &As[0][0], tid);
  stage_tile_async(Wh, bn, kH - 1,  0, &Bs[0][0], tid);
  wait_async_and_barrier();

  int cur = 0;
  for (int k0 = 0; k0 < kH; k0 += kBK) {
    const int nxt = cur ^ 1;
    if (k0 + kBK < kH) {             // async-fill next buffer while we compute
      stage_tile_async(xh, bm, rmax,   k0 + kBK, &As[nxt][0], tid);
      stage_tile_async(Wh, bn, kH - 1, k0 + kBK, &Bs[nxt][0], tid);
    }

    const _Float16* At = &As[cur][0];
    const _Float16* Bt = &Bs[cur][0];

    // Fragments per ISA 7.12.2 layouts
    v16h afrag[2];
#pragma unroll
    for (int mt = 0; mt < 2; ++mt) {
      const int r = wm * 32 + mt * 16 + lo;
      v8h p0 = *(const v8h*)(At + r * kBKP + hi * 8);       // K 0..7  / 8..15
      v8h p1 = *(const v8h*)(At + r * kBKP + hi * 8 + 16);  // K 16..23 / 24..31
      afrag[mt] = combine16(p0, p1);
    }
    v16h bfrag[4];
#pragma unroll
    for (int nt = 0; nt < 4; ++nt) {
      const int c = wn * 64 + nt * 16 + lo;
      v8h p0 = *(const v8h*)(Bt + c * kBKP + hi * 8);       // K 0..7  / 8..15
      v8h p1 = *(const v8h*)(Bt + c * kBKP + hi * 8 + 16);  // K 16..23 / 24..31
      bfrag[nt] = combine16(p0, p1);
    }

#pragma unroll
    for (int mt = 0; mt < 2; ++mt)
#pragma unroll
      for (int nt = 0; nt < 4; ++nt)
        acc[mt][nt] = __builtin_amdgcn_wmma_f32_16x16x32_f16(
            false, afrag[mt], false, bfrag[nt],
            (short)0, acc[mt][nt], false, false);

    wait_async_and_barrier();        // next buffer now resident in LDS
    cur = nxt;
  }

  // Epilogue: +bias, store fp32 y (into d_out; LN pass rewrites in place)
#pragma unroll
  for (int mt = 0; mt < 2; ++mt) {
#pragma unroll
    for (int nt = 0; nt < 4; ++nt) {
      const int gcol = bn + wn * 64 + nt * 16 + lo;
      const float bv = bias[gcol];
#pragma unroll
      for (int r = 0; r < 8; ++r) {
        const int grow = bm + wm * 32 + mt * 16 + r + hi * 8;
        if (grow < M) y[(size_t)grow * kH + gcol] = acc[mt][nt][r] + bv;
      }
    }
  }
}

// per-row: LayerNorm(y)*gamma+beta + x -> RPReLU -> LayerScale (in place, y==out ok)
__global__ __launch_bounds__(256) void ln_rprelu_ls(
    const float* __restrict__ y, const float* __restrict__ x,
    const float* __restrict__ gamma, const float* __restrict__ beta,
    const float* __restrict__ move1, const float* __restrict__ pa,
    const float* __restrict__ move2, const float* __restrict__ lsa,
    const float* __restrict__ lsm, float* __restrict__ out)
{
  const int row   = blockIdx.x;
  const int tid   = threadIdx.x;
  const size_t b0 = (size_t)row * kH;

  float v[3];
#pragma unroll
  for (int i = 0; i < 3; ++i) v[i] = y[b0 + tid + i * 256];

  float s  = v[0] + v[1] + v[2];
  float ss = v[0] * v[0] + v[1] * v[1] + v[2] * v[2];

#pragma unroll
  for (int off = 16; off > 0; off >>= 1) {   // wave32 reduction
    s  += __shfl_down(s, off, 32);
    ss += __shfl_down(ss, off, 32);
  }

  __shared__ float red[16];
  const int wv = tid >> 5, ln = tid & 31;
  if (ln == 0) { red[wv] = s; red[8 + wv] = ss; }
  __syncthreads();
  if (tid == 0) {
    float ts = 0.f, tss = 0.f;
#pragma unroll
    for (int i = 0; i < 8; ++i) { ts += red[i]; tss += red[8 + i]; }
    red[0] = ts; red[8] = tss;
  }
  __syncthreads();

  const float inv  = 1.0f / (float)kH;
  const float mean = red[0] * inv;
  const float var  = red[8] * inv - mean * mean;
  const float rs   = rsqrtf(var + 1e-12f);

#pragma unroll
  for (int i = 0; i < 3; ++i) {
    const int c = tid + i * 256;
    float o = (v[i] - mean) * rs * gamma[c] + beta[c] + x[b0 + c];
    const float z = o - move1[c];
    o = (z >= 0.f) ? z : pa[c] * z;
    o = (o + move2[c]) * lsa[c] + lsm[c];
    out[b0 + c] = o;
  }
}

extern "C" void kernel_launch(void* const* d_in, const int* in_sizes, int n_in,
                              void* d_out, int out_size, void* d_ws, size_t ws_size,
                              hipStream_t stream) {
  const float* x     = (const float*)d_in[0];
  const float* W     = (const float*)d_in[1];
  const float* bias  = (const float*)d_in[2];
  const float* move  = (const float*)d_in[3];
  const float* gamma = (const float*)d_in[4];
  const float* beta  = (const float*)d_in[5];
  const float* move1 = (const float*)d_in[6];
  const float* pa    = (const float*)d_in[7];
  const float* move2 = (const float*)d_in[8];
  const float* lsa   = (const float*)d_in[9];
  const float* lsm   = (const float*)d_in[10];
  float* out = (float*)d_out;

  const int M = in_sizes[0] / kH;            // 64*577 = 36928 rows
  const int totX = M * kH;
  const int totW = kH * kH;

  _Float16* xh = (_Float16*)d_ws;            // [M, 768] f16
  _Float16* Wh = xh + (size_t)totX;          // [768, 768] f16

  cvt_x_f16<<<(totX + 2047) / 2048, 256, 0, stream>>>(x, move, xh, totX);
  cvt_w_f16<<<(totW + 2047) / 2048, 256, 0, stream>>>(W, Wh, totW);

  dim3 ggrid(kH / kBN, (M + kBM - 1) / kBM); // (6, 289)
  gemm_wmma_f16<<<ggrid, 256, 0, stream>>>(xh, Wh, bias, out, M);

  ln_rprelu_ls<<<dim3(M), 256, 0, stream>>>(out, x, gamma, beta, move1, pa,
                                            move2, lsa, lsm, out);
}
